// WindowedAttention_70901320123174
// MI455X (gfx1250) — compile-verified, matrix operand order checked by reading, not run on hardware
//
#include <hip/hip_runtime.h>
#include <hip/hip_bf16.h>

typedef _Float16 half_t;
typedef __attribute__((ext_vector_type(4)))  _Float16 half4v;
typedef __attribute__((ext_vector_type(8)))  _Float16 half8v;
typedef __attribute__((ext_vector_type(16))) _Float16 half16v;
typedef __attribute__((ext_vector_type(8)))  float    float8v;
typedef __attribute__((ext_vector_type(4)))  unsigned int uint32x4;
typedef __attribute__((ext_vector_type(4)))  int      int32x4;
typedef __attribute__((ext_vector_type(8)))  int      int32x8;

#define N_ROWS   50176      // 16 * 3136 rows total (window-ordered)
#define EMB      512
#define WIN_ELE  25088      // 49*512 f16 per window per tensor

#if defined(__has_builtin)
#if __has_builtin(__builtin_amdgcn_tensor_load_to_lds) && \
    __has_builtin(__builtin_amdgcn_s_wait_tensorcnt)
#define HAVE_TDM 1
#endif
#endif

__device__ inline float8v f8zero() {
    float8v v;
#pragma unroll
    for (int i = 0; i < 8; ++i) v[i] = 0.0f;
    return v;
}

__device__ inline half8v h8zero() {
    half8v v;
#pragma unroll
    for (int i = 0; i < 8; ++i) v[i] = (half_t)0.0f;
    return v;
}

// Load one 16-f16 WMMA fragment (A or B operand) from LDS.
// lanes 0-15: K = {0..7} at ptr, {16..23} at ptr+16
// lanes 16-31: K = {8..15} (caller offsets by +8), {24..31} at ptr+16
__device__ inline half16v load_frag(const half_t* ptr) {
    half8v lo = *(const half8v*)ptr;
    half8v hi = *(const half8v*)(ptr + 16);
    return __builtin_shufflevector(lo, hi, 0, 1, 2, 3, 4, 5, 6, 7,
                                           8, 9, 10, 11, 12, 13, 14, 15);
}

__device__ inline float8v wmma_f16(half16v a, half16v b, float8v c) {
    return __builtin_amdgcn_wmma_f32_16x16x32_f16(
        /*neg_a=*/false, a, /*neg_b=*/false, b,
        /*c_mod=*/(short)0, c, /*reuse_a=*/false, /*reuse_b=*/false);
}

#ifdef HAVE_TDM
// One TDM descriptor: 2D f16 tile, 49 rows x 64 elements (128B), row stride
// 64 elements in memory; LDS destination rows padded 128B -> 144B
// (pad_interval code 4 == 32 DWORDs, pad_amount code 3 == 4 DWORDs).
__device__ inline void tdm_load_49x64_f16(const half_t* gsrc, void* lds_dst) {
    const unsigned long long ga  = (unsigned long long)(uintptr_t)gsrc;
    const unsigned int       lds = (unsigned int)(uintptr_t)lds_dst;

    uint32x4 g0;
    g0[0] = 1u;                                   // count=1, user descriptor
    g0[1] = lds;                                  // lds_addr
    g0[2] = (unsigned int)ga;                     // global_addr[31:0]
    g0[3] = (unsigned int)((ga >> 32) & 0x01FFFFFFull)
          | 0x80000000u;                          // global_addr[56:32] | type=2

    int32x8 g1;
    g1[0] = (int)((1u << 16)                      // data_size = 2 bytes
                | (1u << 20)                      // pad_enable
                | (4u << 22)                      // pad_interval: 32 DWORDs
                | (3u << 25));                    // pad_amount: 4 DWORDs
    g1[1] = (int)(64u << 16);                     // tensor_dim0 = 64
    g1[2] = (int)(49u << 16);                     // tensor_dim1 = 49
    g1[3] = (int)(64u << 16);                     // tile_dim0 = 64
    g1[4] = (int)49;                              // tile_dim1 = 49, tile_dim2 = 0
    g1[5] = (int)64;                              // tensor_dim0_stride = 64
    g1[6] = 0;
    g1[7] = 0;

    int32x4 gz4;
    gz4[0] = 0; gz4[1] = 0; gz4[2] = 0; gz4[3] = 0;
    int32x8 gz8;
#pragma unroll
    for (int i = 0; i < 8; ++i) gz8[i] = 0;
    __builtin_amdgcn_tensor_load_to_lds(g0, g1, gz4, gz4, gz8, 0);
}
#endif

// ---------------------------------------------------------------------------
// Phase 1: QKV projection GEMM.  out[g, :] = gather(x)[g, :] @ W^T + b
// Block tile 128x64, K-step 32, 8 waves (wave32); each wave owns a 16x64 strip.
// Output stored window-row-ordered as f16; Q gets the 1/sqrt(d)=0.125 scale.
// ---------------------------------------------------------------------------
__global__ __launch_bounds__(256) void qkv_proj_kernel(
    const float* __restrict__ x,
    const float* __restrict__ qw, const float* __restrict__ qb,
    const float* __restrict__ kw, const float* __restrict__ kb,
    const float* __restrict__ vw, const float* __restrict__ vb,
    half_t* __restrict__ outQ, half_t* __restrict__ outK, half_t* __restrict__ outV)
{
    __shared__ half_t As[128][40];   // 128 rows x 32 k (pad->40 f16, 80B stride)
    __shared__ half_t Bs[64][40];    // 64 out-features x 32 k

    const int tid  = threadIdx.x;
    const int wave = tid >> 5;
    const int lane = tid & 31;
    const int proj = blockIdx.z;

    const float* W;
    const float* bias;
    half_t* out;
    if (proj == 0)      { W = qw; bias = qb; out = outQ; }
    else if (proj == 1) { W = kw; bias = kb; out = outK; }
    else                { W = vw; bias = vb; out = outV; }

    // --- A staging mapping: 2 threads per row, 16 floats each ---
    const int rstage = tid >> 1;           // local row 0..127
    const int rhalf  = (tid & 1) * 16;     // k-subrange within 32
    const int g    = blockIdx.x * 128 + rstage;
    const int b    = g / 3136;
    const int rem  = g - b * 3136;
    const int wIdx = rem / 49;
    const int tkn  = rem - wIdx * 49;
    const int wa = wIdx >> 3, wb = wIdx & 7;
    const int p  = tkn / 7,  q  = tkn - (tkn / 7) * 7;
    const int l  = 392 * wa + 56 * p + 7 * wb + q;
    const float* __restrict__ srcA = x + ((size_t)b * 3136 + l) * EMB + rhalf;

    // --- B staging mapping: 4 threads per weight row, 8 floats each ---
    const int nstage = tid >> 2;           // 0..63
    const int noff   = (tid & 3) * 8;
    const int nBase  = blockIdx.y * 64;
    const float* __restrict__ srcB = W + ((size_t)(nBase + nstage)) * EMB + noff;

    float8v acc[4];
#pragma unroll
    for (int i = 0; i < 4; ++i) acc[i] = f8zero();

    const int kSel = (lane < 16) ? 0 : 8;
    const int arow = wave * 16 + (lane & 15);

    for (int k0 = 0; k0 < EMB; k0 += 32) {
        // Stage A slab (fp32 -> f16)
        {
            const float4* pa = (const float4*)(srcA + k0);
#pragma unroll
            for (int j = 0; j < 4; ++j) {
                float4 v = pa[j];
                half4v hv;
                hv[0] = (half_t)v.x; hv[1] = (half_t)v.y;
                hv[2] = (half_t)v.z; hv[3] = (half_t)v.w;
                *(half4v*)&As[rstage][rhalf + j * 4] = hv;
            }
        }
        // Stage B slab (fp32 -> f16)
        {
            const float4* pb = (const float4*)(srcB + k0);
#pragma unroll
            for (int j = 0; j < 2; ++j) {
                float4 v = pb[j];
                half4v hv;
                hv[0] = (half_t)v.x; hv[1] = (half_t)v.y;
                hv[2] = (half_t)v.z; hv[3] = (half_t)v.w;
                *(half4v*)&Bs[nstage][noff + j * 4] = hv;
            }
        }
        if (k0 + 32 < EMB) {
            // locality 3 -> near-cache (WGP) prefetch of the next K slab
            __builtin_prefetch(srcA + k0 + 32, 0, 3);
            __builtin_prefetch(srcB + k0 + 32, 0, 3);
        }
        __syncthreads();

        half16v afrag = load_frag(&As[arow][kSel]);
#pragma unroll
        for (int tt = 0; tt < 4; ++tt) {
            half16v bfrag = load_frag(&Bs[tt * 16 + (lane & 15)][kSel]);
            acc[tt] = wmma_f16(afrag, bfrag, acc[tt]);
        }
        __syncthreads();
    }

    // Epilogue: bias + (Q scale) + f16 store, window-row-ordered layout.
    const int   mBase = blockIdx.x * 128 + wave * 16 + ((lane < 16) ? 0 : 8);
    const float scale = (proj == 0) ? 0.125f : 1.0f;
#pragma unroll
    for (int tt = 0; tt < 4; ++tt) {
        const int   col = nBase + tt * 16 + (lane & 15);
        const float bv  = bias[col];
#pragma unroll
        for (int i = 0; i < 8; ++i) {
            const int m = mBase + i;
            out[(size_t)m * EMB + col] = (half_t)((acc[tt][i] + bv) * scale);
        }
    }
}

// ---------------------------------------------------------------------------
// Phase 2: per (window, head) attention.  8192 blocks x 128 threads (4 waves).
// Head rows are contiguous: Q/K/V head base = win*25088 + h*3136, rows of 64.
// Q/K/V tiles staged to LDS by the Tensor Data Mover (one 2D descriptor each,
// issued by waves 0/1/2, completion via s_wait_tensorcnt).  49 tokens padded
// to 64; padded key columns masked to -inf before softmax.
// ---------------------------------------------------------------------------
__global__ __launch_bounds__(128) void window_attn_kernel(
    const half_t* __restrict__ Q, const half_t* __restrict__ K,
    const half_t* __restrict__ V,
    const float* __restrict__ bias_table, float* __restrict__ out)
{
    __shared__ half_t Qs[64][72];
    __shared__ half_t Ks[64][72];
    __shared__ half_t Vr[64][72];   // V row-major staging (TDM destination)
    __shared__ half_t Vt[64][72];   // V transposed: [d][key]
    __shared__ half_t Ps[64][72];   // un-normalized exp(scores)
    __shared__ float  Ss[64][65];   // fp32 scores (+bias, masked)
    __shared__ float  rsum[64];     // 1 / softmax denominator

    const int tid  = threadIdx.x;
    const int wave = tid >> 5;
    const int lane = tid & 31;
    const int h    = blockIdx.x & 7;
    const int win  = blockIdx.x >> 3;

    const half_t* __restrict__ Qb = Q + (size_t)win * WIN_ELE + h * 3136;
    const half_t* __restrict__ Kb = K + (size_t)win * WIN_ELE + h * 3136;
    const half_t* __restrict__ Vb = V + (size_t)win * WIN_ELE + h * 3136;

    // --- zero-fill padding (disjoint from TDM-written rows 0..48) ---
    {
        half8v z = h8zero();
        // Qs/Ks rows 49..63: 15 rows x 8 chunks of 8 f16
        for (int i = tid; i < 120; i += 128) {
            const int rr = 49 + (i >> 3);
            const int cc = (i & 7) * 8;
            *(half8v*)&Qs[rr][cc] = z;
            *(half8v*)&Ks[rr][cc] = z;
        }
        // Vt padded key columns 49..63 across all 64 d-rows
        for (int i = tid; i < 960; i += 128) {
            const int dd = i / 15;
            const int kk = 49 + (i - dd * 15);
            Vt[dd][kk] = (half_t)0.0f;
        }
    }

#ifdef HAVE_TDM
    // --- async tile DMA: one descriptor per matrix, one per wave ---
    if (wave == 0) {
        tdm_load_49x64_f16(Qb, &Qs[0][0]);
        __builtin_amdgcn_s_wait_tensorcnt(0);
    } else if (wave == 1) {
        tdm_load_49x64_f16(Kb, &Ks[0][0]);
        __builtin_amdgcn_s_wait_tensorcnt(0);
    } else if (wave == 2) {
        tdm_load_49x64_f16(Vb, &Vr[0][0]);
        __builtin_amdgcn_s_wait_tensorcnt(0);
    }
#else
    {
        const int r     = tid >> 1;          // 0..63
        const int chalf = (tid & 1) * 32;
        if (r < 49) {
            const half8v* sq = (const half8v*)(Qb + r * 64 + chalf);
            const half8v* sk = (const half8v*)(Kb + r * 64 + chalf);
            const half8v* sv = (const half8v*)(Vb + r * 64 + chalf);
#pragma unroll
            for (int j = 0; j < 4; ++j) {
                *(half8v*)&Qs[r][chalf + j * 8] = sq[j];
                *(half8v*)&Ks[r][chalf + j * 8] = sk[j];
                *(half8v*)&Vr[r][chalf + j * 8] = sv[j];
            }
        }
    }
#endif
    __syncthreads();

    const int mrow  = wave * 16 + (lane & 15);
    const int kSel  = (lane < 16) ? 0 : 8;
    const int mBase = wave * 16 + ((lane < 16) ? 0 : 8);

    // --- scores = (Q*0.125) . K^T   (scale folded into Q in phase 1) ---
    float8v sacc[4];
#pragma unroll
    for (int i = 0; i < 4; ++i) sacc[i] = f8zero();
#pragma unroll
    for (int ks = 0; ks < 2; ++ks) {
        half16v aq = load_frag(&Qs[mrow][ks * 32 + kSel]);
#pragma unroll
        for (int tt = 0; tt < 4; ++tt) {
            half16v bk = load_frag(&Ks[tt * 16 + (lane & 15)][ks * 32 + kSel]);
            sacc[tt] = wmma_f16(aq, bk, sacc[tt]);
        }
    }

    // --- + relative position bias, mask padded keys, spill to LDS ---
#pragma unroll
    for (int tt = 0; tt < 4; ++tt) {
        const int col = tt * 16 + (lane & 15);
        const int p2  = col / 7;
        const int q2  = col - p2 * 7;
#pragma unroll
        for (int i = 0; i < 8; ++i) {
            const int m = mBase + i;
            float s = sacc[tt][i];
            if (col < 49) {
                if (m < 49) {
                    const int p1  = m / 7;
                    const int q1  = m - p1 * 7;
                    const int idx = (p1 - p2 + 6) * 13 + (q1 - q2 + 6);
                    s += bias_table[idx * 8 + h];
                }
            } else {
                s = -3.0e30f;   // padded key -> -inf
            }
            Ss[m][col] = s;
        }
    }

    // --- LDS->LDS transpose of V (overlapped with score epilogue) ---
    {
        const int r     = tid >> 1;          // 0..63
        const int chalf = (tid & 1) * 32;
        if (r < 49) {
#pragma unroll
            for (int j = 0; j < 4; ++j) {
                half8v vv = *(const half8v*)&Vr[r][chalf + j * 8];
#pragma unroll
                for (int e = 0; e < 8; ++e) Vt[chalf + j * 8 + e][r] = vv[e];
            }
        }
    }
    __syncthreads();

    // --- row softmax (normalization deferred to epilogue) ---
    if (tid < 64) {
        float mx = -3.0e30f;
        for (int c = 0; c < 64; ++c) mx = fmaxf(mx, Ss[tid][c]);
        float sum = 0.0f;
        for (int c = 0; c < 64; ++c) {
            const float e = __expf(Ss[tid][c] - mx);
            Ps[tid][c] = (half_t)e;
            sum += e;
        }
        rsum[tid] = 1.0f / sum;
    }
    __syncthreads();

    // --- out = softmax(scores) . V ---
    float8v oacc[4];
#pragma unroll
    for (int i = 0; i < 4; ++i) oacc[i] = f8zero();
#pragma unroll
    for (int ks = 0; ks < 2; ++ks) {
        half16v ap = load_frag(&Ps[mrow][ks * 32 + kSel]);
#pragma unroll
        for (int tt = 0; tt < 4; ++tt) {
            half16v bv = load_frag(&Vt[tt * 16 + (lane & 15)][ks * 32 + kSel]);
            oacc[tt] = wmma_f16(ap, bv, oacc[tt]);
        }
    }

    float* __restrict__ outBase = out + (size_t)blockIdx.x * 3136; // (win*8+h)*49*64
#pragma unroll
    for (int tt = 0; tt < 4; ++tt) {
        const int dcol = tt * 16 + (lane & 15);
#pragma unroll
        for (int i = 0; i < 8; ++i) {
            const int m = mBase + i;
            if (m < 49) outBase[m * 64 + dcol] = oacc[tt][i] * rsum[m];
        }
    }
}

extern "C" void kernel_launch(void* const* d_in, const int* in_sizes, int n_in,
                              void* d_out, int out_size, void* d_ws, size_t ws_size,
                              hipStream_t stream) {
    (void)in_sizes; (void)n_in; (void)out_size; (void)ws_size;
    const float* x  = (const float*)d_in[0];
    const float* qw = (const float*)d_in[1];
    const float* qb = (const float*)d_in[2];
    const float* kw = (const float*)d_in[3];
    const float* kb = (const float*)d_in[4];
    const float* vw = (const float*)d_in[5];
    const float* vb = (const float*)d_in[6];
    const float* bt = (const float*)d_in[7];
    float* outp = (float*)d_out;

    // Workspace: Q,K,V as f16, window-row-ordered [50176][512] each (51.4 MB ea).
    half_t* wsQ = (half_t*)d_ws;
    half_t* wsK = wsQ + (size_t)N_ROWS * EMB;
    half_t* wsV = wsK + (size_t)N_ROWS * EMB;

    dim3 g1(N_ROWS / 128, EMB / 64, 3);   // 392 x 8 x 3
    qkv_proj_kernel<<<g1, 256, 0, stream>>>(x, qw, qb, kw, kb, vw, vb,
                                            wsQ, wsK, wsV);
    window_attn_kernel<<<16 * 64 * 8, 128, 0, stream>>>(wsQ, wsK, wsV, bt, outp);
}
